// CstNet2S2_73985106641108
// MI455X (gfx1250) — matrix operand order repaired and checked by reading, not compiled, not run on hardware
//
#include <hip/hip_runtime.h>
#include <math.h>

// ---------------------------------------------------------------------------
// CstNet2S2 forward for MI455X (gfx1250, wave32).
// Dense projections: V_WMMA_F32_16X16X32_F16, branch-free fragment loads.
// All GEMM operands use zero-padded row strides (multiples of 32) so the
// inner loop has NO bounds checks: 4x global_load_b128 (A, f32->f16 cvt) +
// 2x b128 (B, pre-transposed padded f16 weights) + 1 wmma per K-chunk.
// Attention algebra: score_j = g_j.(Wk q) (bk cancels in softmax),
// out = (sum a_j g_j) Wv + bv  -> no k/v materialization.
// Irregular graph ops (kNN, 2-hop dedupe, FPS, softmax) are VALU kernels.
// ---------------------------------------------------------------------------

typedef __attribute__((ext_vector_type(16))) _Float16 v16h;
typedef __attribute__((ext_vector_type(8)))  _Float16 v8h;
typedef __attribute__((ext_vector_type(8)))  float    v8f;

// ---- weight conversion: f32 [Kd,Nd] row-major -> f16 B^T padded [Np][Kp] --
// dst[n*Kp + k] = (k<Kd && n<Nd) ? W[k*Nd + n] : 0
__global__ void cvtT_kernel(const float* __restrict__ src,
                            _Float16* __restrict__ dst,
                            int Kd, int Nd, int Kp, int Np) {
  int i = blockIdx.x * blockDim.x + threadIdx.x;
  if (i >= Kp * Np) return;
  int n = i / Kp, k = i % Kp;
  float v = (k < Kd && n < Nd) ? src[(size_t)k * Nd + n] : 0.0f;
  dst[i] = (_Float16)v;
}
// ---- plain padded copy: f32 [C,C] row-major -> f16 [Cp][Cp] ---------------
// used for Wk so that t = q @ Wk^T reads contiguously: B(k,n)=Wk[n,k]
__global__ void cvtP_kernel(const float* __restrict__ src,
                            _Float16* __restrict__ dst, int C, int Cp) {
  int i = blockIdx.x * blockDim.x + threadIdx.x;
  if (i >= Cp * Cp) return;
  int n = i / Cp, k = i % Cp;
  float v = (k < C && n < C) ? src[(size_t)n * C + k] : 0.0f;
  dst[i] = (_Float16)v;
}

// ---- WMMA GEMM: Y[M,Np] = act(X[M,Kp] @ B (+bias)) ------------------------
// X: f32, row stride Kp (mult of 32, zero-padded). Bh: f16 B^T [Np][Kp].
// M multiple of 16. Pad columns [Nd,Np) of Y are written as zeros.
// block = (32,4): 4 waves, each owns one 16x16 output tile along M.
__global__ void gemm_wmma_kernel(const float* __restrict__ X,
                                 const _Float16* __restrict__ Bh,
                                 const float* __restrict__ bias,
                                 float* __restrict__ Y,
                                 int M, int Kp, int Nd, int Np, int relu) {
  union AF { v16h v; v8h g[2]; _Float16 h[16]; };
  union CF { v8f v; float f[8]; };
  const int lane = threadIdx.x;
  const int tileM = blockIdx.x * 4 + threadIdx.y;
  const int m0 = tileM * 16;
  if (m0 >= M) return;                       // wave-uniform exit
  const int n0 = blockIdx.y * 16;
  const int half = (lane >= 16) ? 1 : 0;
  const int rowA = m0 + (lane & 15);
  const int colB = n0 + (lane & 15);

  const float* xr = X + (size_t)rowA * Kp + half * 8;
  const _Float16* wr = Bh + (size_t)colB * Kp + half * 8;

  CF acc;
#pragma unroll
  for (int j = 0; j < 8; ++j) acc.f[j] = 0.0f;

  for (int k0 = 0; k0 < Kp; k0 += 32) {
    float4 x0 = *(const float4*)(xr + k0);
    float4 x1 = *(const float4*)(xr + k0 + 4);
    float4 x2 = *(const float4*)(xr + k0 + 16);
    float4 x3 = *(const float4*)(xr + k0 + 20);
    AF a, b;
    a.h[0] = (_Float16)x0.x;  a.h[1] = (_Float16)x0.y;
    a.h[2] = (_Float16)x0.z;  a.h[3] = (_Float16)x0.w;
    a.h[4] = (_Float16)x1.x;  a.h[5] = (_Float16)x1.y;
    a.h[6] = (_Float16)x1.z;  a.h[7] = (_Float16)x1.w;
    a.h[8] = (_Float16)x2.x;  a.h[9] = (_Float16)x2.y;
    a.h[10] = (_Float16)x2.z; a.h[11] = (_Float16)x2.w;
    a.h[12] = (_Float16)x3.x; a.h[13] = (_Float16)x3.y;
    a.h[14] = (_Float16)x3.z; a.h[15] = (_Float16)x3.w;
    b.g[0] = *(const v8h*)(wr + k0);
    b.g[1] = *(const v8h*)(wr + k0 + 16);
    acc.v = __builtin_amdgcn_wmma_f32_16x16x32_f16(
        false, a.v, false, b.v, (short)0, acc.v, false, false);
  }

  // C/D layout: VGPR j -> row m0 + j + 8*half, col = n0 + (lane&15)
  int cb = colB < Nd ? colB : 0;
  float bv = bias ? bias[cb] : 0.0f;
  bool valid = (colB < Nd);
#pragma unroll
  for (int j = 0; j < 8; ++j) {
    int row = m0 + j + half * 8;
    float v = acc.f[j] + bv;
    if (relu && v < 0.0f) v = 0.0f;
    Y[(size_t)row * Np + colB] = valid ? v : 0.0f;
  }
}

// ---------------- tiny dense layer (bandwidth-bound, VALU) -----------------
__global__ void linear_kernel(const float* __restrict__ X,
                              const float* __restrict__ W,
                              const float* __restrict__ bias,
                              float* __restrict__ Y,
                              int M, int Ci, int Co,
                              int ldx, int offx, int ldy, int offy, int relu) {
  int gid = blockIdx.x * blockDim.x + threadIdx.x;
  if (gid >= M * Co) return;
  int m = gid / Co, co = gid % Co;
  const float* xr = X + (size_t)m * ldx + offx;
  float acc = bias[co];
  for (int c = 0; c < Ci; ++c) acc += xr[c] * W[(size_t)c * Co + co];
  if (relu && acc < 0.0f) acc = 0.0f;
  Y[(size_t)m * ldy + offy + co] = acc;
}

// ---------------- strided slice copy ---------------------------------------
__global__ void copy_slice_kernel(const float* __restrict__ X,
                                  float* __restrict__ Y,
                                  int M, int C, int ldx, int offx,
                                  int ldy, int offy) {
  int gid = blockIdx.x * blockDim.x + threadIdx.x;
  if (gid >= M * C) return;
  int m = gid / C, c = gid % C;
  Y[(size_t)m * ldy + offy + c] = X[(size_t)m * ldx + offx + c];
}

// ---------------- 11-NN (includes self), one thread per point --------------
__global__ void knn11_kernel(const float* __restrict__ x, int* __restrict__ nb,
                             int B, int N, int C) {
  int gid = blockIdx.x * blockDim.x + threadIdx.x;
  if (gid >= B * N) return;
  int b = gid / N, n = gid % N;
  const float* xb = x + (size_t)b * N * C;
  const float* xn = xb + (size_t)n * C;
  float bd[12]; int bi[12];
#pragma unroll
  for (int j = 0; j < 12; ++j) { bd[j] = 1e30f; bi[j] = 0; }
  for (int m = 0; m < N; ++m) {
    const float* xm = xb + (size_t)m * C;
    float d = 0.0f;
    for (int c = 0; c < C; ++c) { float t = xn[c] - xm[c]; d += t * t; }
    if (d < bd[10]) {
      bd[11] = d; bi[11] = m;
#pragma unroll
      for (int j = 10; j >= 0; --j) {
        if (bd[j + 1] < bd[j]) {
          float td = bd[j]; bd[j] = bd[j + 1]; bd[j + 1] = td;
          int ti = bi[j]; bi[j] = bi[j + 1]; bi[j + 1] = ti;
        }
      }
    }
  }
  for (int j = 0; j < 11; ++j) nb[(size_t)gid * 11 + j] = bi[j];
}

// ---------------- surface_knn: 2-hop candidates, dedupe, top-32 ------------
#define TH_PTS 32
__global__ void twohop_kernel(const float* __restrict__ x,
                              const int* __restrict__ nb,
                              int* __restrict__ idx_all, int B, int N, int C) {
  __shared__ int   cS[TH_PTS][121];
  __shared__ float dS[TH_PTS][122];
  int t = threadIdx.x;
  int gid = blockIdx.x * TH_PTS + t;
  if (gid >= B * N) return;
  int b = gid / N, n = gid % N;
  int* cand = cS[t];
  float* dd = dS[t];
  const int* nbn = nb + (size_t)gid * 11;
  for (int a = 0; a < 11; ++a) {
    int na = nbn[a];
    const int* nba = nb + ((size_t)b * N + na) * 11;
    for (int c2 = 0; c2 < 11; ++c2) cand[a * 11 + c2] = nba[c2];
  }
  const float* xb = x + (size_t)b * N * C;
  const float* xn = xb + (size_t)n * C;
  for (int j = 0; j < 121; ++j) {
    const float* xm = xb + (size_t)cand[j] * C;
    float d = 0.0f;
    for (int c = 0; c < C; ++c) { float q = xn[c] - xm[c]; d += q * q; }
    dd[j] = d;
  }
  for (int j = 1; j < 121; ++j) {   // mask duplicates (keep first occurrence)
    int cj = cand[j]; bool dup = false;
    for (int i = 0; i < j; ++i) dup |= (cand[i] == cj);
    if (dup) dd[j] = 1e30f;
  }
  int* out = idx_all + (size_t)gid * 32;
  for (int r = 0; r < 32; ++r) {    // 32 x argmin over 121
    float best = dd[0]; int bj = 0;
    for (int j = 1; j < 121; ++j) if (dd[j] < best) { best = dd[j]; bj = j; }
    out[r] = cand[bj]; dd[bj] = 1e30f;
  }
}

// ---------------- farthest point sampling (one block per batch) ------------
__global__ void fps_kernel(const float* __restrict__ x, int* __restrict__ fps,
                           int B, int N, int C, int S) {
  __shared__ float dist[2048];
  __shared__ float xp[128];
  __shared__ float rv[256];
  __shared__ int   ri[256];
  __shared__ int   lastS;
  int b = blockIdx.x, t = threadIdx.x, nt = blockDim.x;
  const float* xb = x + (size_t)b * N * C;
  for (int n = t; n < N; n += nt) dist[n] = 1e30f;
  if (t == 0) { fps[(size_t)b * S] = 0; lastS = 0; }
  __syncthreads();
  for (int i = 1; i < S; ++i) {
    int last = lastS;
    for (int c = t; c < C; c += nt) xp[c] = xb[(size_t)last * C + c];
    __syncthreads();
    float bestd = -1.0f; int bestn = N;
    for (int n = t; n < N; n += nt) {
      const float* xm = xb + (size_t)n * C;
      float d = 0.0f;
      for (int c = 0; c < C; ++c) { float q = xm[c] - xp[c]; d += q * q; }
      float nd = dist[n] < d ? dist[n] : d;
      dist[n] = nd;
      if (nd > bestd || (nd == bestd && n < bestn)) { bestd = nd; bestn = n; }
    }
    rv[t] = bestd; ri[t] = bestn;
    __syncthreads();
    for (int off = nt >> 1; off > 0; off >>= 1) {
      if (t < off) {
        if (rv[t + off] > rv[t] ||
            (rv[t + off] == rv[t] && ri[t + off] < ri[t])) {
          rv[t] = rv[t + off]; ri[t] = ri[t + off];
        }
      }
      __syncthreads();
    }
    if (t == 0) { fps[(size_t)b * S + i] = ri[0]; lastS = ri[0]; }
    __syncthreads();
  }
}

// ---------------- gathers (padded output rows, zero-filled pads) -----------
__global__ void gather_rows_kernel(const float* __restrict__ X,
                                   const int* __restrict__ ind,
                                   float* __restrict__ Y,
                                   int BS, int S, int Nin, int C, int Cp) {
  int gid = blockIdx.x * blockDim.x + threadIdx.x;
  if (gid >= BS * Cp) return;
  int bs = gid / Cp, c = gid % Cp;
  int b = bs / S;
  Y[gid] = (c < C) ? X[((size_t)b * Nin + ind[bs]) * C + c] : 0.0f;
}

__global__ void gather_idx_kernel(const int* __restrict__ idx_all,
                                  const int* __restrict__ fps,
                                  int* __restrict__ idx,
                                  int BS, int S, int Nin) {
  int gid = blockIdx.x * blockDim.x + threadIdx.x;
  if (gid >= BS * 32) return;
  int bs = gid / 32, k = gid % 32;
  int b = bs / S;
  idx[gid] = idx_all[((size_t)b * Nin + fps[bs]) * 32 + k];
}

// ---------------- point attention core: scores/softmax/weighted-sum --------
// One wave per (b,s). score_j = scale * g_j . t ; gbar = sum a_j g_j.
// t has row stride ldT; gbar written with row stride ldG (pads zeroed).
__global__ void attn_core_kernel(const float* __restrict__ stream,
                                 const int* __restrict__ idx,
                                 const float* __restrict__ t,
                                 float* __restrict__ gbar,
                                 int S, int Nin, int C, int ldT, int ldG,
                                 float scale) {
  __shared__ float aS[32];
  __shared__ int   iS[32];
  int bs = blockIdx.x;
  int lane = threadIdx.x;
  int b = bs / S;
  const float* tv = t + (size_t)bs * ldT;
  int ind = idx[(size_t)bs * 32 + lane];
  const float* g = stream + ((size_t)b * Nin + ind) * C;
  float sc = 0.0f;
  for (int c = 0; c < C; ++c) sc += g[c] * tv[c];
  sc *= scale;
  float m = sc;
#pragma unroll
  for (int off = 16; off > 0; off >>= 1) {
    float o = __shfl_xor(m, off, 32); m = o > m ? o : m;
  }
  float e = __expf(sc - m);
  float s = e;
#pragma unroll
  for (int off = 16; off > 0; off >>= 1) s += __shfl_xor(s, off, 32);
  aS[lane] = e / s; iS[lane] = ind;
  __syncthreads();
  for (int c = lane; c < ldG; c += 32) {
    float acc = 0.0f;
    if (c < C) {
#pragma unroll 4
      for (int j = 0; j < 32; ++j)
        acc += aS[j] * stream[((size_t)b * Nin + iS[j]) * C + c];
    }
    gbar[(size_t)bs * ldG + c] = acc;
  }
}

// ---------------- feature attention core (K=7, contiguous stack) -----------
__global__ void fea_core_kernel(const float* __restrict__ stk,
                                const float* __restrict__ t,
                                float* __restrict__ gbar,
                                int C, float scale) {
  __shared__ float aS[8];
  int bs = blockIdx.x, lane = threadIdx.x;
  const float* tv = t + (size_t)bs * C;
  if (lane < 7) {
    const float* g = stk + ((size_t)bs * 7 + lane) * C;
    float sc = 0.0f;
    for (int c = 0; c < C; ++c) sc += g[c] * tv[c];
    aS[lane] = sc * scale;
  }
  __syncthreads();
  if (lane == 0) {
    float m = aS[0];
    for (int j = 1; j < 7; ++j) m = aS[j] > m ? aS[j] : m;
    float s = 0.0f;
    for (int j = 0; j < 7; ++j) { float e = __expf(aS[j] - m); aS[j] = e; s += e; }
    for (int j = 0; j < 7; ++j) aS[j] /= s;
  }
  __syncthreads();
  for (int c = lane; c < C; c += 32) {
    float acc = 0.0f;
    for (int j = 0; j < 7; ++j) acc += aS[j] * stk[((size_t)bs * 7 + j) * C + c];
    gbar[(size_t)bs * C + c] = acc;
  }
}

// ---------------- global max pool + log-softmax ----------------------------
__global__ void maxpool_kernel(const float* __restrict__ x,
                               float* __restrict__ y, int B, int S, int C) {
  int gid = blockIdx.x * blockDim.x + threadIdx.x;
  if (gid >= B * C) return;
  int b = gid / C, c = gid % C;
  float m = -1e30f;
  for (int s = 0; s < S; ++s) {
    float v = x[((size_t)b * S + s) * C + c];
    m = v > m ? v : m;
  }
  y[gid] = m;
}

__global__ void logsoftmax_kernel(const float* __restrict__ x,
                                  float* __restrict__ y, int B, int C, int ld) {
  int b = blockIdx.x * blockDim.x + threadIdx.x;
  if (b >= B) return;
  const float* xr = x + (size_t)b * ld;
  float m = -1e30f;
  for (int c = 0; c < C; ++c) m = xr[c] > m ? xr[c] : m;
  float s = 0.0f;
  for (int c = 0; c < C; ++c) s += __expf(xr[c] - m);
  float ls = __logf(s);
  for (int c = 0; c < C; ++c) y[(size_t)b * C + c] = xr[c] - m - ls;
}

// ===========================================================================
// Host orchestration
// ===========================================================================
extern "C" void kernel_launch(void* const* d_in, const int* in_sizes, int n_in,
                              void* d_out, int out_size, void* d_ws,
                              size_t ws_size, hipStream_t stream) {
  (void)in_sizes; (void)n_in; (void)out_size; (void)ws_size;
  const int B = 16, N = 2048;
  const int NinL[3]  = {2048, 1024, 512};
  const int SL[3]    = {1024, 512, 256};
  const int cinL[3]  = {8, 32, 128};
  const int coutL[3] = {32, 128, 256};
  const int THR = 256;
  auto rup32 = [](int v) { return (v + 31) & ~31; };

  // ---- parameter pointer walk (recursive insertion-order flattening) ----
  // d_in: 0..5 = xyz,pmt,mad,dim,nor,loc ; then params.proj (7 mlps x 2 lin
  // x {W,b}), params.ssa (3 x [attn 7x{q,k,v}x{W,b}, mlp 7x2x{W,b},
  // fea {q,k,v}x{W,b}]), params.head (2x{W,b}).
  int cur = 6;
  auto F = [&]() -> const float* { return (const float*)d_in[cur++]; };
  struct LinP { const float* W; const float* b; };
  LinP proj[7][2];
  LinP aQ[3][7], aK[3][7], aV[3][7], sM[3][7][2], fQ[3], fK[3], fV[3], head[2];
  for (int i = 0; i < 7; ++i)
    for (int l = 0; l < 2; ++l) { proj[i][l].W = F(); proj[i][l].b = F(); }
  for (int lv = 0; lv < 3; ++lv) {
    for (int s = 0; s < 7; ++s) {
      aQ[lv][s].W = F(); aQ[lv][s].b = F();
      aK[lv][s].W = F(); aK[lv][s].b = F();
      aV[lv][s].W = F(); aV[lv][s].b = F();
    }
    for (int s = 0; s < 7; ++s)
      for (int l = 0; l < 2; ++l) { sM[lv][s][l].W = F(); sM[lv][s][l].b = F(); }
    fQ[lv].W = F(); fQ[lv].b = F();
    fK[lv].W = F(); fK[lv].b = F();
    fV[lv].W = F(); fV[lv].b = F();
  }
  head[0].W = F(); head[0].b = F();
  head[1].W = F(); head[1].b = F();

  // ---- workspace bump allocator ----
  char* wsp = (char*)d_ws;
  size_t off = 0;
  auto alloc = [&](size_t bytes) -> void* {
    void* p = wsp + off;
    off = (off + bytes + 255) & ~(size_t)255;
    return p;
  };
  const size_t SB = (size_t)16 * 65536;               // max stream elems
  float* sbuf[15];
  for (int i = 0; i < 15; ++i) sbuf[i] = (float*)alloc(SB * 4);
  float* sin[7];  float* sout[7];  float* spare = sbuf[14];
  for (int i = 0; i < 7; ++i) { sin[i] = sbuf[i]; sout[i] = sbuf[7 + i]; }
  float* concat = (float*)alloc((size_t)B * N * 48 * 4);
  float* hid    = (float*)alloc((size_t)B * N * 8  * 4);
  float* csth   = (float*)alloc((size_t)B * N * 24 * 4);
  int*   nb     = (int*)alloc((size_t)B * N * 11 * 4);
  int*   idxall = (int*)alloc((size_t)B * N * 32 * 4);
  int*   fpsi   = (int*)alloc((size_t)B * 1024 * 4);
  int*   idx    = (int*)alloc((size_t)B * 1024 * 32 * 4);
  // padded attention scratch: stride = rup32(cin), max is 16*1024*32 floats
  float* center = (float*)alloc((size_t)B * 1024 * 128 * 4);
  float* qb     = (float*)alloc((size_t)B * 1024 * 128 * 4);
  float* tb     = (float*)alloc((size_t)B * 1024 * 128 * 4);
  float* gbar   = (float*)alloc((size_t)B * 1024 * 128 * 4);
  float* aout   = (float*)alloc((size_t)B * 1024 * 128 * 4);
  float* mlph   = (float*)alloc((size_t)B * 256 * 256 * 4 * 2);  // max B*S*cout
  float* stk    = (float*)alloc((size_t)B * 256 * 7 * 256 * 4);
  float* feaq   = (float*)alloc((size_t)B * 256 * 256 * 4 * 2);
  float* feat_  = (float*)alloc((size_t)B * 256 * 256 * 4 * 2);
  float* feag   = (float*)alloc((size_t)B * 256 * 256 * 4 * 2);
  float* pooled = (float*)alloc((size_t)B * 256 * 4);
  float* headh  = (float*)alloc((size_t)B * 128 * 4);
  float* logits = (float*)alloc((size_t)B * 48 * 4);

  // weight converters into padded B^T (cvtT) / padded plain (cvtP) f16
  auto cvtT = [&](const float* w, int Kd, int Nd) -> const _Float16* {
    int Kp = rup32(Kd), Np = (Nd + 15) & ~15;
    int n = Kp * Np;
    _Float16* d = (_Float16*)alloc((size_t)n * 2);
    cvtT_kernel<<<(n + THR - 1) / THR, THR, 0, stream>>>(w, d, Kd, Nd, Kp, Np);
    return d;
  };
  auto cvtP = [&](const float* w, int C) -> const _Float16* {
    int Cp = rup32(C);
    int n = Cp * Cp;
    _Float16* d = (_Float16*)alloc((size_t)n * 2);
    cvtP_kernel<<<(n + THR - 1) / THR, THR, 0, stream>>>(w, d, C, Cp);
    return d;
  };
  auto gemm = [&](const float* X, const _Float16* W, const float* bias,
                  float* Y, int M, int Kp, int Nd, int Np, int relu) {
    dim3 g((M / 16 + 3) / 4, (Np + 15) / 16);
    dim3 blk(32, 4);
    gemm_wmma_kernel<<<g, blk, 0, stream>>>(X, W, bias, Y, M, Kp, Nd, Np, relu);
  };

  // f16 copies of all WMMA-consumed weights (recomputed each call)
  const _Float16 *q16[3][7], *k16[3][7], *v16[3][7], *m116[3][7], *m216[3][7];
  const _Float16 *fq16[3], *fk16[3], *fv16[3];
  for (int lv = 0; lv < 3; ++lv) {
    int ci = cinL[lv], co = coutL[lv];
    for (int s = 0; s < 7; ++s) {
      q16[lv][s]  = cvtT(aQ[lv][s].W, ci, ci);
      k16[lv][s]  = cvtP(aK[lv][s].W, ci);        // used as B^T of Wk^T
      v16[lv][s]  = cvtT(aV[lv][s].W, ci, ci);
      m116[lv][s] = cvtT(sM[lv][s][0].W, ci, co);
      m216[lv][s] = cvtT(sM[lv][s][1].W, co, co);
    }
    fq16[lv] = cvtT(fQ[lv].W, co, co);
    fk16[lv] = cvtP(fK[lv].W, co);
    fv16[lv] = cvtT(fV[lv].W, co, co);
  }
  const _Float16* h016 = cvtT(head[0].W, 256, 128);
  const _Float16* h116 = cvtT(head[1].W, 128, 40);

  // ---- input projections (tiny channels -> VALU linear kernels) ----
  const int ciIn[6] = {3, 5, 3, 1, 3, 3};
  const int M0 = B * N;
  for (int i = 0; i < 6; ++i) {
    const float* x = (const float*)d_in[i];
    linear_kernel<<<(M0 * 8 + THR - 1) / THR, THR, 0, stream>>>(
        x, proj[i][0].W, proj[i][0].b, hid, M0, ciIn[i], 8, ciIn[i], 0, 8, 0, 1);
    linear_kernel<<<(M0 * 8 + THR - 1) / THR, THR, 0, stream>>>(
        hid, proj[i][1].W, proj[i][1].b, sin[i], M0, 8, 8, 8, 0, 8, 0, 0);
    copy_slice_kernel<<<(M0 * 8 + THR - 1) / THR, THR, 0, stream>>>(
        sin[i], concat, M0, 8, 8, 0, 48, 8 * i);
  }
  linear_kernel<<<(M0 * 24 + THR - 1) / THR, THR, 0, stream>>>(
      concat, proj[6][0].W, proj[6][0].b, csth, M0, 48, 24, 48, 0, 24, 0, 1);
  linear_kernel<<<(M0 * 8 + THR - 1) / THR, THR, 0, stream>>>(
      csth, proj[6][1].W, proj[6][1].b, sin[6], M0, 24, 8, 24, 0, 8, 0, 0);

  // ---- 3 MultiSSA levels ----
  for (int lv = 0; lv < 3; ++lv) {
    int Nin = NinL[lv], S = SL[lv], ci = cinL[lv], co = coutL[lv];
    int cip = rup32(ci);
    int BS = B * S;
    float scaleP = 1.0f / sqrtf((float)ci);
    float scaleF = 1.0f / sqrtf((float)co);

    // surface_knn + FPS on xyz feature stream
    knn11_kernel<<<(B * Nin + THR - 1) / THR, THR, 0, stream>>>(
        sin[0], nb, B, Nin, ci);
    twohop_kernel<<<(B * Nin + TH_PTS - 1) / TH_PTS, TH_PTS, 0, stream>>>(
        sin[0], nb, idxall, B, Nin, ci);
    fps_kernel<<<B, 256, 0, stream>>>(sin[0], fpsi, B, Nin, ci, S);
    gather_idx_kernel<<<(BS * 32 + THR - 1) / THR, THR, 0, stream>>>(
        idxall, fpsi, idx, BS, S, Nin);

    for (int s = 0; s < 7; ++s) {
      // center gather + q = center@Wq+bq ; t = q@Wk^T (bk cancels in softmax)
      gather_rows_kernel<<<(BS * cip + THR - 1) / THR, THR, 0, stream>>>(
          sin[s], fpsi, center, BS, S, Nin, ci, cip);
      gemm(center, q16[lv][s], aQ[lv][s].b, qb, BS, cip, ci, cip, 0);
      gemm(qb, k16[lv][s], nullptr, tb, BS, cip, ci, cip, 0);
      attn_core_kernel<<<BS, 32, 0, stream>>>(sin[s], idx, tb, gbar,
                                              S, Nin, ci, cip, cip, scaleP);
      gemm(gbar, v16[lv][s], aV[lv][s].b, aout, BS, cip, ci, cip, 0);
      // per-stream MLP: ci -> co (relu) -> co   (co is a multiple of 32)
      gemm(aout, m116[lv][s], sM[lv][s][0].b, mlph, BS, cip, co, co, 1);
      gemm(mlph, m216[lv][s], sM[lv][s][1].b, sout[s], BS, co, co, co, 0);
    }

    // feature attention over the 7 new streams (cst stream is the query)
    for (int m = 0; m < 7; ++m)
      copy_slice_kernel<<<(BS * co + THR - 1) / THR, THR, 0, stream>>>(
          sout[m], stk, BS, co, co, 0, 7 * co, m * co);
    gemm(sout[6], fq16[lv], fQ[lv].b, feaq, BS, co, co, co, 0);
    gemm(feaq, fk16[lv], nullptr, feat_, BS, co, co, co, 0);
    fea_core_kernel<<<BS, 32, 0, stream>>>(stk, feat_, feag, co, scaleF);
    gemm(feag, fv16[lv], fV[lv].b, spare, BS, co, co, co, 0);
    { float* old = sout[6]; sout[6] = spare; spare = old; }

    // rotate stream double-buffers
    for (int i = 0; i < 7; ++i) { float* t2 = sin[i]; sin[i] = sout[i]; sout[i] = t2; }
  }

  // ---- global max pool over 256 points -> head MLP -> log_softmax ----
  maxpool_kernel<<<(B * 256 + THR - 1) / THR, THR, 0, stream>>>(
      sin[6], pooled, B, 256, 256);
  gemm(pooled, h016, head[0].b, headh, B, 256, 128, 128, 1);
  gemm(headh, h116, head[1].b, logits, B, 128, 40, 48, 0);
  logsoftmax_kernel<<<1, 64, 0, stream>>>(logits, (float*)d_out, B, 40, 48);
}
